// SelfAttentionNarrow_62715112456860
// MI455X (gfx1250) — matrix operand order repaired and checked by reading, not compiled, not run on hardware
//
#include <hip/hip_runtime.h>
#include <hip/hip_bf16.h>
#include <math.h>

// ---------------------------------------------------------------------------
// SelfAttentionNarrow on MI455X (gfx1250), wave32 + WMMA f32_16x16x32_f16.
// Flash-attention formulation (2048x2048 scores never hit HBM).
// K/V tiles staged into LDS by the Tensor Data Mover (double buffered,
// TENSORcnt-synchronized) so DMA overlaps WMMA + softmax.
// ---------------------------------------------------------------------------

typedef __attribute__((ext_vector_type(16))) _Float16     v16h;
typedef __attribute__((ext_vector_type(8)))  float        v8f;
typedef __attribute__((ext_vector_type(4)))  float        v4f;
typedef __attribute__((ext_vector_type(4)))  unsigned int v4u;

typedef __attribute__((ext_vector_type(4))) unsigned int tdm_g0_t;
typedef __attribute__((ext_vector_type(8))) int          tdm_g1_t;
typedef __attribute__((ext_vector_type(4))) int          tdm_g2_t;
typedef __attribute__((ext_vector_type(8))) int          tdm_g4_t;

#define B_ 4
#define T_ 2048
#define E_ 1024
#define H_ 16
#define S_ 64

#if __has_builtin(__builtin_amdgcn_tensor_load_to_lds)
#define USE_TDM 1
#endif

union ABu { v16h h; v4u u[2]; };

__device__ __forceinline__ v8f wmma_f16(v16h a, v16h b, v8f c) {
  return __builtin_amdgcn_wmma_f32_16x16x32_f16(false, a, false, b,
                                                (short)0, c, false, false);
}

// A operand (16x32 f16, MxK): lane = row (lane&15); halves 0..7 = K 8g..8g+7,
// halves 8..15 = K 16+8g..23+8g (g = lane>>4), per ISA 16-bit A table.
__device__ __forceinline__ v16h load_a_f16(const _Float16* base, int ld) {
  const int lane = threadIdx.x & 31;
  const int r = lane & 15, g = lane >> 4;
  const _Float16* p = base + (size_t)r * ld + 8 * g;
  ABu t;
  t.u[0] = *(const v4u*)p;
  t.u[1] = *(const v4u*)(p + 16);
  return t.h;
}

__device__ __forceinline__ v16h load_a_f32cvt(const float* base, int ld) {
  const int lane = threadIdx.x & 31;
  const int r = lane & 15, g = lane >> 4;
  const float* p0 = base + (size_t)r * ld + 8 * g;
  const float* p1 = p0 + 16;
  v4f f0 = ((const v4f*)p0)[0], f1 = ((const v4f*)p0)[1];
  v4f f2 = ((const v4f*)p1)[0], f3 = ((const v4f*)p1)[1];
  v16h o;
#pragma unroll
  for (int i = 0; i < 4; ++i) {
    o[i]      = (_Float16)f0[i];
    o[4 + i]  = (_Float16)f1[i];
    o[8 + i]  = (_Float16)f2[i];
    o[12 + i] = (_Float16)f3[i];
  }
  return o;
}

// B operand (32x16 f16, KxN) from [N,K] row-major: lane = col n (lane&15);
// halves 0..7 = K 8g..8g+7, halves 8..15 = K 16+8g..23+8g (matches A).
__device__ __forceinline__ v16h load_b_f16(const _Float16* baseNK, int ld) {
  const int lane = threadIdx.x & 31;
  const int n = lane & 15, g = lane >> 4;
  const _Float16* p = baseNK + (size_t)n * ld + 8 * g;
  ABu t;
  t.u[0] = *(const v4u*)p;
  t.u[1] = *(const v4u*)(p + 16);
  return t.h;
}

#ifdef USE_TDM
// Issue one TDM 2-D tile load (f16 elements) into LDS.  D# per ISA ch.8:
// group0 = {count=1, lds_addr, global_addr[56:0], type=2}
// group1 = {data_size=1(2B), tensor_dim=tile_dim (no OOB), tile dims, stride0}
__device__ __forceinline__ void tdm_load_2d(unsigned lds_byte_off,
                                            const _Float16* gptr,
                                            unsigned d0, unsigned d1,
                                            unsigned long long stride0) {
  unsigned long long ga = (unsigned long long)(uintptr_t)gptr;
  tdm_g0_t g0;
  g0[0] = 1u;                                        // count=1, user mode
  g0[1] = lds_byte_off;                              // lds_addr (bytes)
  g0[2] = (unsigned)(ga & 0xFFFFFFFFu);              // global_addr[31:0]
  g0[3] = (unsigned)((ga >> 32) & 0x1FFFFFFu) | (2u << 30);  // [56:32]|type=2
  tdm_g1_t g1;
  g1[0] = (int)(1u << 16);                           // data_size=1 (2 bytes)
  g1[1] = (int)((d0 & 0xFFFFu) << 16);               // tensor_dim0[15:0]<<16
  g1[2] = (int)(((d0 >> 16) & 0xFFFFu) | ((d1 & 0xFFFFu) << 16));
  g1[3] = (int)(((d1 >> 16) & 0xFFFFu) | ((d0 & 0xFFFFu) << 16)); // tile_dim0
  g1[4] = (int)(d1 & 0xFFFFu);                       // tile_dim1, tile_dim2=0
  g1[5] = (int)(unsigned)(stride0 & 0xFFFFFFFFu);    // tensor_dim0_stride lo
  g1[6] = (int)(unsigned)((stride0 >> 32) & 0xFFFFu);
  g1[7] = 0;
  tdm_g2_t z4 = {0, 0, 0, 0};
#if defined(__clang_major__) && (__clang_major__ >= 23)
  tdm_g4_t z8 = {0, 0, 0, 0, 0, 0, 0, 0};
  __builtin_amdgcn_tensor_load_to_lds(g0, g1, z4, z4, z8, 0);
#else
  __builtin_amdgcn_tensor_load_to_lds(g0, g1, z4, z4, 0);
#endif
}
#endif  // USE_TDM

// ---------------------------------------------------------------------------
// Kernel 0a/0b: weight conversion (Wu -> f16; Wq/Wk/Wv -> f16 with the
// 1024^-0.25 attention scale folded into Wq and Wk).
// ---------------------------------------------------------------------------
__global__ void cvt_wu_kernel(const float* __restrict__ Wu,
                              _Float16* __restrict__ WuH, int n) {
  int i = blockIdx.x * blockDim.x + threadIdx.x;
  if (i < n) WuH[i] = (_Float16)Wu[i];
}

__global__ void cvt_w3_kernel(const float* __restrict__ Wq,
                              const float* __restrict__ Wk,
                              const float* __restrict__ Wv,
                              _Float16* __restrict__ WqH,
                              _Float16* __restrict__ WkH,
                              _Float16* __restrict__ WvH) {
  int i = blockIdx.x * blockDim.x + threadIdx.x;  // 4096 total
  const float qk = 0.17677669529663687f;          // 1024^-0.25
  WqH[i] = (_Float16)(Wq[i] * qk);
  WkH[i] = (_Float16)(Wk[i] * qk);
  WvH[i] = (_Float16)Wv[i];
}

// ---------------------------------------------------------------------------
// Kernel 1: QKV projection.  One wave per (b,h, 16-row t-tile).
// ---------------------------------------------------------------------------
__device__ __forceinline__ void do_proj(v16h a0, v16h a1,
                                        const _Float16* __restrict__ W,
                                        _Float16* __restrict__ dst,
                                        size_t row_stride, size_t col_stride,
                                        int t0) {
  const int lane = threadIdx.x & 31;
  const int r = lane & 15, g = lane >> 4;
#pragma unroll
  for (int nt = 0; nt < 4; ++nt) {
    v8f c = {};
    c = wmma_f16(a0, load_b_f16(W + nt * 16 * S_, S_), c);
    c = wmma_f16(a1, load_b_f16(W + nt * 16 * S_ + 32, S_), c);
    _Float16 hv[8];
#pragma unroll
    for (int i = 0; i < 8; ++i) hv[i] = (_Float16)c[i];  // distinct regs
#pragma unroll
    for (int i = 0; i < 8; ++i) {
      size_t row = (size_t)t0 + i + 8 * g;     // t
      size_t col = (size_t)nt * 16 + r;        // s
      dst[row * row_stride + col * col_stride] = hv[i];
    }
  }
}

__global__ __launch_bounds__(128) void qkv_proj_kernel(
    const float* __restrict__ x, const _Float16* __restrict__ WqH,
    const _Float16* __restrict__ WkH, const _Float16* __restrict__ WvH,
    _Float16* __restrict__ Qh, _Float16* __restrict__ Kh,
    _Float16* __restrict__ Vt) {
  const int wave = threadIdx.x >> 5;
  const int tile = blockIdx.x * 4 + wave;     // 8192 tiles
  const int b  = tile >> 11;
  const int h  = (tile >> 7) & 15;
  const int t0 = (tile & 127) << 4;
  const int bh = b * H_ + h;

  const float* xbase = x + ((size_t)b * T_ + t0) * E_ + h * S_;
  v16h a0 = load_a_f32cvt(xbase, E_);
  v16h a1 = load_a_f32cvt(xbase + 32, E_);

  do_proj(a0, a1, WqH, Qh + (size_t)bh * T_ * S_, S_, 1, t0);
  do_proj(a0, a1, WkH, Kh + (size_t)bh * T_ * S_, S_, 1, t0);
  do_proj(a0, a1, WvH, Vt + (size_t)bh * S_ * T_, 1, T_, t0);  // transposed
}

// ---------------------------------------------------------------------------
// Kernel 2: flash attention.  One wave per (b*h, 16-query tile); 64 kv-tiles
// of 32.  K/V tiles double-buffered into LDS via TDM; online softmax row
// stats per D-VGPR; P relayouted D->A through per-wave LDS staging.
// ---------------------------------------------------------------------------
#define ATT_WAVES 2

__global__ __launch_bounds__(32 * ATT_WAVES) void flash_attn_kernel(
    const _Float16* __restrict__ Qh, const _Float16* __restrict__ Kh,
    const _Float16* __restrict__ Vt, _Float16* __restrict__ AO) {
#ifdef USE_TDM
  // [wave][buf][0]=K tile 32x64, [wave][buf][1]=V tile 64x32 (4KB each)
  __shared__ __align__(16) _Float16 kv_lds[ATT_WAVES][2][2][2048];
#endif
  __shared__ __align__(16) _Float16 lds_p[ATT_WAVES][16 * 40];

  const int wave = threadIdx.x >> 5;
  const int lane = threadIdx.x & 31;
  const int tile = blockIdx.x * ATT_WAVES + wave;   // 8192
  const int bh = tile >> 7;
  const int qt = tile & 127;
  const int r = lane & 15, g = lane >> 4;

  const _Float16* qbase = Qh + ((size_t)bh * T_ + qt * 16) * S_;
  const v16h qa0 = load_a_f16(qbase, S_);
  const v16h qa1 = load_a_f16(qbase + 32, S_);

  float m[8], l[8];
#pragma unroll
  for (int i = 0; i < 8; ++i) { m[i] = -3.0e38f; l[i] = 0.0f; }
  v8f O0 = {}, O1 = {}, O2 = {}, O3 = {};

  _Float16* sp = &lds_p[wave][0];
  const _Float16* kmat = Kh + (size_t)bh * T_ * S_;
  const _Float16* vmat = Vt + (size_t)bh * S_ * T_;

#ifdef USE_TDM
  // prologue: DMA tile 0 into buffer 0
  tdm_load_2d((unsigned)(uintptr_t)&kv_lds[wave][0][0][0], kmat, 64, 32, 64);
  tdm_load_2d((unsigned)(uintptr_t)&kv_lds[wave][0][1][0], vmat, 32, 64, T_);
#endif

#pragma unroll 1
  for (int kt = 0; kt < 64; ++kt) {
    const int kb = kt * 32;
#ifdef USE_TDM
    if (kt + 1 < 64) {  // DMA next tile into other buffer, overlap compute
      tdm_load_2d((unsigned)(uintptr_t)&kv_lds[wave][(kt + 1) & 1][0][0],
                  kmat + (size_t)(kb + 32) * S_, 64, 32, 64);
      tdm_load_2d((unsigned)(uintptr_t)&kv_lds[wave][(kt + 1) & 1][1][0],
                  vmat + (kb + 32), 32, 64, T_);
      __builtin_amdgcn_s_wait_tensorcnt(2);   // current buffer complete
    } else {
      __builtin_amdgcn_s_wait_tensorcnt(0);
    }
    asm volatile("" ::: "memory");
    const _Float16* kt_l = &kv_lds[wave][kt & 1][0][0];  // 32 x 64
    const _Float16* vt_l = &kv_lds[wave][kt & 1][1][0];  // 64 x 32
    const int kld = 64, vld = 32;
    const _Float16* kb0 = kt_l;
    const _Float16* kb1 = kt_l + 16 * 64;
#else
    const _Float16* kb0 = kmat + (size_t)kb * S_;
    const _Float16* kb1 = kmat + (size_t)(kb + 16) * S_;
    const _Float16* vt_l = vmat + kb;
    const int kld = S_, vld = T_;
    if (kt + 1 < 64)
      __builtin_prefetch((const char*)(kmat + (size_t)(kb + 32) * S_) +
                             (threadIdx.x & 31) * 128, 0, 1);
#endif
    // ---- S = Q K^T for 32 keys
    v8f S0 = {}, S1 = {};
    S0 = wmma_f16(qa0, load_b_f16(kb0, kld), S0);
    S0 = wmma_f16(qa1, load_b_f16(kb0 + 32, kld), S0);
    S1 = wmma_f16(qa0, load_b_f16(kb1, kld), S1);
    S1 = wmma_f16(qa1, load_b_f16(kb1 + 32, kld), S1);

    // ---- online softmax (row = i + 8*g, cols striped over 16 lanes)
#pragma unroll
    for (int i = 0; i < 8; ++i) {
      float mx = fmaxf(S0[i], S1[i]);
      mx = fmaxf(mx, __shfl_xor(mx, 1));
      mx = fmaxf(mx, __shfl_xor(mx, 2));
      mx = fmaxf(mx, __shfl_xor(mx, 4));
      mx = fmaxf(mx, __shfl_xor(mx, 8));
      const float mn  = fmaxf(m[i], mx);
      const float fac = __expf(m[i] - mn);
      m[i] = mn;
      const float p0 = __expf(S0[i] - mn);
      const float p1 = __expf(S1[i] - mn);
      float sum = p0 + p1;
      sum += __shfl_xor(sum, 1);
      sum += __shfl_xor(sum, 2);
      sum += __shfl_xor(sum, 4);
      sum += __shfl_xor(sum, 8);
      l[i] = l[i] * fac + sum;
      O0[i] *= fac; O1[i] *= fac; O2[i] *= fac; O3[i] *= fac;
      sp[(i + 8 * g) * 40 + r]      = (_Float16)p0;   // stage P, D layout
      sp[(i + 8 * g) * 40 + 16 + r] = (_Float16)p1;
    }

    asm volatile("s_wait_dscnt 0" ::: "memory");  // same-wave DS RAW

    ABu pt;   // reload P in A-operand layout (M=16, K=32)
    pt.u[0] = *(const v4u*)(sp + r * 40 + 8 * g);
    pt.u[1] = *(const v4u*)(sp + r * 40 + 16 + 8 * g);
    const v16h pa = pt.h;

    // ---- O += P V : B operand = Vt rows ([N=s, K=t])
    O0 = wmma_f16(pa, load_b_f16(vt_l + (size_t)0  * vld, vld), O0);
    O1 = wmma_f16(pa, load_b_f16(vt_l + (size_t)16 * vld, vld), O1);
    O2 = wmma_f16(pa, load_b_f16(vt_l + (size_t)32 * vld, vld), O2);
    O3 = wmma_f16(pa, load_b_f16(vt_l + (size_t)48 * vld, vld), O3);
  }

  // ---- epilogue: O / l -> AO f16 concat layout [b, t, h*64+s]
  const int b = bh >> 4, h = bh & 15;
#pragma unroll
  for (int i = 0; i < 8; ++i) {
    const float inv = 1.0f / l[i];
    const size_t row = (size_t)b * T_ + qt * 16 + i + 8 * g;
    _Float16* o = AO + row * E_ + h * S_;
    _Float16 h0 = (_Float16)(O0[i] * inv), h1 = (_Float16)(O1[i] * inv);
    _Float16 h2 = (_Float16)(O2[i] * inv), h3 = (_Float16)(O3[i] * inv);
    o[r] = h0; o[16 + r] = h1; o[32 + r] = h2; o[48 + r] = h3;
  }
}

// ---------------------------------------------------------------------------
// Kernel 3: output projection  out = AO @ Wu^T + bu   (8192 x 1024 x 1024)
// ---------------------------------------------------------------------------
__global__ __launch_bounds__(128) void out_proj_kernel(
    const _Float16* __restrict__ AO, const _Float16* __restrict__ WuH,
    const float* __restrict__ bu, float* __restrict__ out) {
  const int wave = threadIdx.x >> 5;
  const int lane = threadIdx.x & 31;
  const int tile = blockIdx.x * 4 + wave;   // 8192
  const int mt = tile >> 4;
  const int n0 = (tile & 15) * 64;
  const int r = lane & 15, g = lane >> 4;

  const _Float16* abase = AO + (size_t)mt * 16 * E_;
  v8f c0 = {}, c1 = {}, c2 = {}, c3 = {};
#pragma unroll 4
  for (int ks = 0; ks < 32; ++ks) {
    const v16h a = load_a_f16(abase + ks * 32, E_);
    c0 = wmma_f16(a, load_b_f16(WuH + (size_t)(n0 + 0)  * E_ + ks * 32, E_), c0);
    c1 = wmma_f16(a, load_b_f16(WuH + (size_t)(n0 + 16) * E_ + ks * 32, E_), c1);
    c2 = wmma_f16(a, load_b_f16(WuH + (size_t)(n0 + 32) * E_ + ks * 32, E_), c2);
    c3 = wmma_f16(a, load_b_f16(WuH + (size_t)(n0 + 48) * E_ + ks * 32, E_), c3);
  }
  const float b0 = bu[n0 + r],      b1 = bu[n0 + 16 + r];
  const float b2 = bu[n0 + 32 + r], b3 = bu[n0 + 48 + r];
#pragma unroll
  for (int i = 0; i < 8; ++i) {
    const size_t row = (size_t)mt * 16 + i + 8 * g;
    float* o = out + row * E_ + n0;
    float f0 = c0[i] + b0, f1 = c1[i] + b1, f2 = c2[i] + b2, f3 = c3[i] + b3;
    o[r] = f0; o[16 + r] = f1; o[32 + r] = f2; o[48 + r] = f3;
  }
}

// ---------------------------------------------------------------------------
extern "C" void kernel_launch(void* const* d_in, const int* in_sizes, int n_in,
                              void* d_out, int out_size, void* d_ws,
                              size_t ws_size, hipStream_t stream) {
  const float* x  = (const float*)d_in[0];
  const float* Wk = (const float*)d_in[1];
  const float* Wq = (const float*)d_in[2];
  const float* Wv = (const float*)d_in[3];
  const float* Wu = (const float*)d_in[4];
  const float* bu = (const float*)d_in[5];
  float* out = (float*)d_out;

  const size_t QKV_ELEMS = (size_t)B_ * H_ * T_ * S_;   // 8,388,608
  _Float16* Qh  = (_Float16*)d_ws;
  _Float16* Kh  = Qh + QKV_ELEMS;
  _Float16* Vt  = Kh + QKV_ELEMS;
  _Float16* AO  = Vt + QKV_ELEMS;                       // [b, t, e] f16
  _Float16* WuH = AO + QKV_ELEMS;                       // 1 Mi halves
  _Float16* WqH = WuH + (size_t)E_ * E_;                // 4 Ki halves each
  _Float16* WkH = WqH + S_ * S_;
  _Float16* WvH = WkH + S_ * S_;

  cvt_wu_kernel<<<(E_ * E_) / 256, 256, 0, stream>>>(Wu, WuH, E_ * E_);
  cvt_w3_kernel<<<(S_ * S_) / 256, 256, 0, stream>>>(Wq, Wk, Wv, WqH, WkH, WvH);
  qkv_proj_kernel<<<2048, 128, 0, stream>>>(x, WqH, WkH, WvH, Qh, Kh, Vt);
  flash_attn_kernel<<<8192 / ATT_WAVES, 32 * ATT_WAVES, 0, stream>>>(Qh, Kh,
                                                                     Vt, AO);
  out_proj_kernel<<<2048, 128, 0, stream>>>(AO, WuH, bu, out);
}
